// TransformerBlock_58471684767878
// MI455X (gfx1250) — matrix operand order repaired and checked by reading, not compiled, not run on hardware
//
#include <hip/hip_runtime.h>
#include <math.h>

// ---------------- problem constants ----------------
#define BB   2
#define SS   2048
#define HH   2048
#define NHH  32
#define KVHH 8
#define DD   64
#define QPKK 4
#define KVV  512
#define FFF  8192
#define MM   (BB*SS)        // 4096 rows

typedef __bf16 bf16_t;
typedef __bf16 v4bf  __attribute__((ext_vector_type(4)));
typedef __bf16 v8bf  __attribute__((ext_vector_type(8)));
typedef __bf16 v16bf __attribute__((ext_vector_type(16)));
typedef float  v8f   __attribute__((ext_vector_type(8)));

union FragAB { v16bf v; v8bf h[2]; bf16_t e[16]; };
union FragC  { v8f   v; float  f[8]; };

static __device__ __forceinline__ v8f wmma_bf16(v16bf a, v16bf b, v8f c) {
    // D(f32 16x16) = A(bf16 16x32) * B(bf16 32x16) + C
    return __builtin_amdgcn_wmma_f32_16x16x32_bf16(false, a, false, b,
                                                   (short)0, c, false, false);
}

static __device__ __forceinline__ v8bf ldb8(const bf16_t* p) {
    return *(const v8bf*)p;
}

// exchange value with lane^16 (SWAPX16 ds_swizzle: xor=0x10, and=0x1f)
static __device__ __forceinline__ float swz16(float x) {
    int r = __builtin_amdgcn_ds_swizzle((int)__float_as_uint(x), 0x401F);
    return __uint_as_float((unsigned)r);
}

// ---------------- weight convert + transpose: W[K x N] f32 -> Wt[N x K] bf16 ----
// 32x32 tile per block via LDS so both global read and write are coalesced.
__global__ __launch_bounds__(256) void wt_kernel(const float* __restrict__ w,
                                                 bf16_t* __restrict__ wt,
                                                 int K, int N) {
    __shared__ bf16_t tile[32][33];
    const int n0 = blockIdx.x * 32, k0 = blockIdx.y * 32;
    const int tx = threadIdx.x & 31, ty = threadIdx.x >> 5;   // 32 x 8
#pragma unroll
    for (int i = 0; i < 4; ++i) {
        int r = ty + 8 * i;
        tile[r][tx] = (bf16_t)w[(long)(k0 + r) * N + n0 + tx];
    }
    __syncthreads();
#pragma unroll
    for (int i = 0; i < 4; ++i) {
        int r = ty + 8 * i;
        wt[(long)(n0 + r) * K + k0 + tx] = tile[tx][r];
    }
}

// ---------------- LayerNorm (one row of H per block), bf16 out --------------
__global__ __launch_bounds__(256) void ln_kernel(const float* __restrict__ x,
                                                 const float* __restrict__ g,
                                                 const float* __restrict__ be,
                                                 bf16_t* __restrict__ out) {
    const int row = blockIdx.x;
    const float4* xr = (const float4*)(x + (long)row * HH);
    float4 v4[2];
    float s = 0.f, s2 = 0.f;
#pragma unroll
    for (int i = 0; i < 2; ++i) {
        float4 v = xr[threadIdx.x + 256 * i];
        v4[i] = v;
        s  += v.x + v.y + v.z + v.w;
        s2 += v.x * v.x + v.y * v.y + v.z * v.z + v.w * v.w;
    }
    __shared__ float rs[256], rq[256];
    rs[threadIdx.x] = s; rq[threadIdx.x] = s2;
    __syncthreads();
    for (int off = 128; off > 0; off >>= 1) {
        if (threadIdx.x < off) {
            rs[threadIdx.x] += rs[threadIdx.x + off];
            rq[threadIdx.x] += rq[threadIdx.x + off];
        }
        __syncthreads();
    }
    const float mu  = rs[0] * (1.f / HH);
    const float var = rq[0] * (1.f / HH) - mu * mu;
    const float inv = rsqrtf(var + 1e-5f);
#pragma unroll
    for (int i = 0; i < 2; ++i) {
        int c = (threadIdx.x + 256 * i) * 4;
        const float4 gv = *(const float4*)(g + c);
        const float4 bv = *(const float4*)(be + c);
        v4bf o;
        o[0] = (bf16_t)((v4[i].x - mu) * inv * gv.x + bv.x);
        o[1] = (bf16_t)((v4[i].y - mu) * inv * gv.y + bv.y);
        o[2] = (bf16_t)((v4[i].z - mu) * inv * gv.z + bv.z);
        o[3] = (bf16_t)((v4[i].w - mu) * inv * gv.w + bv.w);
        *(v4bf*)(out + (long)row * HH + c) = o;
    }
}

// ---------------- generic WMMA GEMM: out = epi(A[MxK]bf16 @ Wt[NxK]^T + bias) ----
// 256 threads = 8 waves; workgroup tile 128(M) x 128(N).
// Wave: 32(M) x 64(N) = 2 M-tiles x 4 N-tiles -> 8 WMMAs per K=32 step.
template<bool GELU_ACT, bool HAS_RES, bool OUT_F32, bool OUT_BF16, bool VT_MODE>
__global__ __launch_bounds__(256) void gemm_kernel(
    const bf16_t* __restrict__ A, const bf16_t* __restrict__ Wt,
    const float*  __restrict__ bias, const float* __restrict__ res,
    float* __restrict__ outf, bf16_t* __restrict__ outb,
    int Mdim, int Ndim, int Kdim)
{
    const int tid  = threadIdx.x;
    const int lane = tid & 31, wv = tid >> 5;
    const int half = lane >> 4, nq = lane & 15;
    const int m0 = blockIdx.y * 128 + 32 * (wv & 3);
    const int n0 = blockIdx.x * 128 + 64 * (wv >> 2);

    long arow[2], brow[4];
#pragma unroll
    for (int mi = 0; mi < 2; ++mi) arow[mi] = (long)(m0 + 16 * mi + nq) * Kdim;
#pragma unroll
    for (int ni = 0; ni < 4; ++ni) brow[ni] = (long)(n0 + 16 * ni + nq) * Kdim;

    FragC c[2][4];
#pragma unroll
    for (int mi = 0; mi < 2; ++mi)
#pragma unroll
        for (int ni = 0; ni < 4; ++ni)
#pragma unroll
            for (int i = 0; i < 8; ++i) c[mi][ni].f[i] = 0.f;

#pragma unroll 2
    for (int kk = 0; kk < Kdim; kk += 32) {
        __builtin_prefetch((const void*)(A + arow[0] + kk + 64), 0, 1);
        FragAB a[2], bf[4];
#pragma unroll
        for (int mi = 0; mi < 2; ++mi) {
            // A fragment: rows m=nq, K = kk + 16*half + 0..15 (contiguous)
            a[mi].h[0] = ldb8(A + arow[mi] + kk + 16 * half);
            a[mi].h[1] = ldb8(A + arow[mi] + kk + 16 * half + 8);
        }
#pragma unroll
        for (int ni = 0; ni < 4; ++ni) {
            // B fragment: col n=nq, K = kk + 16*half + 0..15 (contiguous)
            bf[ni].h[0] = ldb8(Wt + brow[ni] + kk + 16 * half);
            bf[ni].h[1] = ldb8(Wt + brow[ni] + kk + 16 * half + 8);
        }
#pragma unroll
        for (int mi = 0; mi < 2; ++mi)
#pragma unroll
            for (int ni = 0; ni < 4; ++ni)
                c[mi][ni].v = wmma_bf16(a[mi].v, bf[ni].v, c[mi][ni].v);
    }

    float bn[4];
#pragma unroll
    for (int ni = 0; ni < 4; ++ni) bn[ni] = bias[n0 + 16 * ni + nq];

#pragma unroll
    for (int mi = 0; mi < 2; ++mi)
#pragma unroll
        for (int r = 0; r < 8; ++r) {
            const int m = m0 + 16 * mi + r + 8 * half;
#pragma unroll
            for (int ni = 0; ni < 4; ++ni) {
                const int n = n0 + 16 * ni + nq;
                float v = c[mi][ni].f[r] + bn[ni];
                if (GELU_ACT) v = 0.5f * v * (1.f + erff(v * 0.70710678118f));
                if (HAS_RES)  v += res[(long)m * Ndim + n];
                if (OUT_F32)  outf[(long)m * Ndim + n] = v;
                if (OUT_BF16) {
                    if (VT_MODE) {
                        // scatter V projection to vT[b][kvh][d][s]
                        const int bi = m >> 11, s = m & (SS - 1);
                        const int kvh = n >> 6, d = n & 63;
                        outb[(((long)(bi * KVHH + kvh) * DD) + d) * SS + s] =
                            (bf16_t)v;
                    } else {
                        outb[(long)m * Ndim + n] = (bf16_t)v;
                    }
                }
            }
        }
}

// ---------------- flash attention: one wave per (b, head, 16-query tile) ------
// scoresT = K_tile * Q_tile^T  (softmax stats are per-lane: lane owns query nq)
// oT     += vT_tile * pT       (pT built in-register from scoresT + swz16)
__global__ __launch_bounds__(256) void attn_kernel(
    const bf16_t* __restrict__ qb,   // [B*S, H]    bf16
    const bf16_t* __restrict__ kb,   // [B*S, KV]   bf16
    const bf16_t* __restrict__ vtb,  // [B,KVH,D,S] bf16
    const float*  __restrict__ mask, // [B, S]
    bf16_t* __restrict__ ob)         // [B*S, H]    bf16
{
    const int tid  = threadIdx.x;
    const int lane = tid & 31, wvl = tid >> 5;
    const int half = lane >> 4, nq = lane & 15;
    const int wid  = blockIdx.x * 8 + wvl;
    const int qt  = wid & 127;          // S/16 tiles
    const int h   = (wid >> 7) & (NHH - 1);
    const int b   = wid >> 12;
    const int kvh = h >> 2;             // QPK = 4

    // Q as B-operand: d = {8*half.., 16+8*half..}, per K=32 chunk (d<32, d>=32)
    const bf16_t* qp = qb + ((long)(b * SS + qt * 16 + nq)) * HH + h * DD;
    FragAB qf0, qf1;
    qf0.h[0] = ldb8(qp + 8 * half);       qf0.h[1] = ldb8(qp + 16 + 8 * half);
    qf1.h[0] = ldb8(qp + 32 + 8 * half);  qf1.h[1] = ldb8(qp + 48 + 8 * half);

    FragC oacc[4];
#pragma unroll
    for (int mt = 0; mt < 4; ++mt)
#pragma unroll
        for (int r = 0; r < 8; ++r) oacc[mt].f[r] = 0.f;

    float m_run = -1e30f, l_run = 0.f;
    const float scale = 0.125f; // 1/sqrt(64)

    const bf16_t* kbp = kb + (long)b * SS * KVV + kvh * DD;
    const bf16_t* vbp = vtb + ((long)(b * KVHH + kvh)) * DD * SS;
    const float*  mrw = mask + (long)b * SS;

    for (int kc = 0; kc < SS; kc += 32) {
        // ---- scoresT: two 16x16 tiles (k rows kc..kc+15, kc+16..kc+31) ----
        FragC sc[2];
#pragma unroll
        for (int t = 0; t < 2; ++t) {
            const bf16_t* kp = kbp + (long)(kc + 16 * t + nq) * KVV;
            FragAB a0, a1;
            a0.h[0] = ldb8(kp + 8 * half);       a0.h[1] = ldb8(kp + 16 + 8 * half);
            a1.h[0] = ldb8(kp + 32 + 8 * half);  a1.h[1] = ldb8(kp + 48 + 8 * half);
            FragC z;
#pragma unroll
            for (int i = 0; i < 8; ++i) z.f[i] = 0.f;
            z.v = wmma_bf16(a0.v, qf0.v, z.v);
            sc[t].v = wmma_bf16(a1.v, qf1.v, z.v);
        }
        // ---- scale + mask + online softmax over k (per-lane query nq) ----
        float e[16];
        float cmax = -1e30f;
#pragma unroll
        for (int t = 0; t < 2; ++t) {
            const float4* mp = (const float4*)(mrw + kc + 16 * t + 8 * half);
            float4 ma = mp[0], mb = mp[1];
            float mv[8] = {ma.x, ma.y, ma.z, ma.w, mb.x, mb.y, mb.z, mb.w};
#pragma unroll
            for (int r = 0; r < 8; ++r) {
                float sv = sc[t].f[r] * scale + mv[r];
                e[t * 8 + r] = sv;
                cmax = fmaxf(cmax, sv);
            }
        }
        cmax = fmaxf(cmax, swz16(cmax));
        const float newm  = fmaxf(m_run, cmax);
        const float alpha = __expf(m_run - newm);
        float ssum = 0.f;
#pragma unroll
        for (int i = 0; i < 16; ++i) { e[i] = __expf(e[i] - newm); ssum += e[i]; }
        ssum += swz16(ssum);
        l_run = l_run * alpha + ssum;
        m_run = newm;
#pragma unroll
        for (int mt = 0; mt < 4; ++mt)
#pragma unroll
            for (int r = 0; r < 8; ++r) oacc[mt].f[r] *= alpha;

        // ---- build pT as B-fragment: lane needs K = 16*half .. 16*half+15 ----
        FragAB pf;
#pragma unroll
        for (int j = 0; j < 8; ++j) {
            float ex0 = swz16(e[j]);       // partner's k = j + 8*(1-half)      (tile 0)
            float ex1 = swz16(e[8 + j]);   // partner's k = 16 + j + 8*(1-half) (tile 1)
            pf.e[j]     = (bf16_t)(half ? ex1 : e[j]);
            pf.e[8 + j] = (bf16_t)(half ? e[8 + j] : ex0);
        }
        // ---- oT += vT_tile(16d x 32k) * pT ----
#pragma unroll
        for (int mt = 0; mt < 4; ++mt) {
            const bf16_t* vp = vbp + (long)(16 * mt + nq) * SS + kc;
            FragAB av;
            av.h[0] = ldb8(vp + 16 * half);
            av.h[1] = ldb8(vp + 16 * half + 8);
            oacc[mt].v = wmma_bf16(av.v, pf.v, oacc[mt].v);
        }
    }

    const float inv = 1.f / l_run;
    const long obase = ((long)(b * SS + qt * 16 + nq)) * HH + h * DD;
#pragma unroll
    for (int mt = 0; mt < 4; ++mt)
#pragma unroll
        for (int r = 0; r < 8; ++r) {
            const int d = 16 * mt + r + 8 * half;
            ob[obase + d] = (bf16_t)(oacc[mt].f[r] * inv);
        }
}

// --------------------------------------------------------------------------
extern "C" void kernel_launch(void* const* d_in, const int* in_sizes, int n_in,
                              void* d_out, int out_size, void* d_ws, size_t ws_size,
                              hipStream_t stream) {
    const float* x   = (const float*)d_in[0];
    const float* msk = (const float*)d_in[1];
    const float* Wq  = (const float*)d_in[2];  const float* bq = (const float*)d_in[3];
    const float* Wk  = (const float*)d_in[4];  const float* bk = (const float*)d_in[5];
    const float* Wv  = (const float*)d_in[6];  const float* bv = (const float*)d_in[7];
    const float* Wo  = (const float*)d_in[8];  const float* bo = (const float*)d_in[9];
    const float* W1  = (const float*)d_in[10]; const float* b1 = (const float*)d_in[11];
    const float* W2  = (const float*)d_in[12]; const float* b2 = (const float*)d_in[13];
    const float* g1  = (const float*)d_in[14]; const float* be1 = (const float*)d_in[15];
    const float* g2  = (const float*)d_in[16]; const float* be2 = (const float*)d_in[17];

    char* w = (char*)d_ws;
    size_t off = 0;
    auto alloc = [&](size_t bytes) -> void* {
        void* p = w + off;
        off += (bytes + 255) & ~(size_t)255;
        return p;
    };
    bf16_t* Wtq = (bf16_t*)alloc((size_t)HH * HH * 2);
    bf16_t* Wtk = (bf16_t*)alloc((size_t)HH * KVV * 2);
    bf16_t* Wtv = (bf16_t*)alloc((size_t)HH * KVV * 2);
    bf16_t* Wto = (bf16_t*)alloc((size_t)HH * HH * 2);
    bf16_t* Wt1 = (bf16_t*)alloc((size_t)HH * FFF * 2);
    bf16_t* Wt2 = (bf16_t*)alloc((size_t)FFF * HH * 2);
    bf16_t* h1   = (bf16_t*)alloc((size_t)MM * HH * 2);
    bf16_t* qbf  = (bf16_t*)alloc((size_t)MM * HH * 2);
    bf16_t* kbf  = (bf16_t*)alloc((size_t)MM * KVV * 2);
    bf16_t* vtbf = (bf16_t*)alloc((size_t)MM * KVV * 2);
    bf16_t* atbf = (bf16_t*)alloc((size_t)MM * HH * 2);
    float*  x2   = (float*)alloc((size_t)MM * HH * 4);
    bf16_t* h2   = (bf16_t*)alloc((size_t)MM * HH * 2);
    bf16_t* m1   = (bf16_t*)alloc((size_t)MM * FFF * 2);

    const dim3 blk(256);

    // weight convert + transpose to bf16 [N x K] (coalesced via LDS tiles)
    wt_kernel<<<dim3(HH / 32, HH / 32), blk, 0, stream>>>(Wq, Wtq, HH, HH);
    wt_kernel<<<dim3(KVV / 32, HH / 32), blk, 0, stream>>>(Wk, Wtk, HH, KVV);
    wt_kernel<<<dim3(KVV / 32, HH / 32), blk, 0, stream>>>(Wv, Wtv, HH, KVV);
    wt_kernel<<<dim3(HH / 32, HH / 32), blk, 0, stream>>>(Wo, Wto, HH, HH);
    wt_kernel<<<dim3(FFF / 32, HH / 32), blk, 0, stream>>>(W1, Wt1, HH, FFF);
    wt_kernel<<<dim3(HH / 32, FFF / 32), blk, 0, stream>>>(W2, Wt2, FFF, HH);

    // h1 = LN(x)
    ln_kernel<<<MM, blk, 0, stream>>>(x, g1, be1, h1);

    // Q/K/V projections (WMMA)
    gemm_kernel<false, false, false, true, false>
        <<<dim3(HH / 128, MM / 128), blk, 0, stream>>>(h1, Wtq, bq, nullptr,
                                                       nullptr, qbf, MM, HH, HH);
    gemm_kernel<false, false, false, true, false>
        <<<dim3(KVV / 128, MM / 128), blk, 0, stream>>>(h1, Wtk, bk, nullptr,
                                                        nullptr, kbf, MM, KVV, HH);
    gemm_kernel<false, false, false, true, true>
        <<<dim3(KVV / 128, MM / 128), blk, 0, stream>>>(h1, Wtv, bv, nullptr,
                                                        nullptr, vtbf, MM, KVV, HH);

    // flash attention (WMMA scores + AV)
    attn_kernel<<<(BB * NHH * (SS / 16)) / 8, blk, 0, stream>>>(qbf, kbf, vtbf,
                                                                msk, atbf);

    // x2 = x + attn @ Wo + bo
    gemm_kernel<false, true, true, false, false>
        <<<dim3(HH / 128, MM / 128), blk, 0, stream>>>(atbf, Wto, bo, x,
                                                       x2, nullptr, MM, HH, HH);

    // h2 = LN(x2)
    ln_kernel<<<MM, blk, 0, stream>>>(x2, g2, be2, h2);

    // m1 = gelu(h2 @ W1 + b1)
    gemm_kernel<true, false, false, true, false>
        <<<dim3(FFF / 128, MM / 128), blk, 0, stream>>>(h2, Wt1, b1, nullptr,
                                                        nullptr, m1, MM, FFF, HH);

    // out = x2 + m1 @ W2 + b2
    gemm_kernel<false, true, true, false, false>
        <<<dim3(HH / 128, MM / 128), blk, 0, stream>>>(m1, Wt2, b2, x2,
                                                       (float*)d_out, nullptr,
                                                       MM, HH, FFF);
}
